// seg_model_11699490915025
// MI455X (gfx1250) — compile-verified, matrix operand order checked by reading, not run on hardware
//
#include <hip/hip_runtime.h>

// ---------------------------------------------------------------------------
// ChebNet forward for MI455X (gfx1250). All heavy matmuls run through
// v_wmma_f32_16x16x32_bf16 (bf16 in, f32 accumulate). L (8x2048x2048 bf16 =
// 64MB) is L2-resident (192MB L2), so the 11 L@X passes stream from L2.
// Weights are pre-transposed once into bf16 [Nout][K] fragment-major layout
// (zero-padded), so GEMM staging is unconditional 16B chunks. Tile staging
// uses the CDNA5 async copy path (global_load_async_to_lds_b128 + ASYNCcnt);
// LDS tiles are fragment-major ([row][K], stride 40 elems = 80B) so every
// 16x32 WMMA fragment is two 16B ds_load_b128 per lane.
// ---------------------------------------------------------------------------

typedef __attribute__((ext_vector_type(16))) __bf16 v16bf;
typedef __attribute__((ext_vector_type(8)))  __bf16 v8bf;
typedef __attribute__((ext_vector_type(8)))  float  v8f;

#define NPTS 2048
#define NB   8
#define ROWS (NB * NPTS)   // 16384

// Block tile 128(M) x 128(N), K-step 32. 8 waves arranged 4x2; each wave owns
// 32x64 = 2x4 fragments -> 8 WMMAs per K-step per wave.
#define BM 128
#define BN 128
#define BK 32
#define LST 40   // LDS row stride in bf16 elems (80B: 16B-aligned chunks)

// CDNA5 async copy: per-lane 16B global -> LDS, tracked by ASYNCcnt.
#define ASYNC_CP16(ldsaddr, gptr)                                            \
    asm volatile("global_load_async_to_lds_b128 %0, %1, off"                 \
                 :: "v"(ldsaddr), "v"((unsigned long long)(size_t)(gptr))    \
                 : "memory")
#define ASYNC_DRAIN() asm volatile("s_wait_asynccnt 0x0" ::: "memory")

// -------------------------- Laplacian build (VALU) -------------------------

__global__ void sq_kernel(const float* __restrict__ x, float* __restrict__ sq) {
    int i = blockIdx.x * 256 + threadIdx.x;          // 0..16383
    const float* p = x + (size_t)i * 6;
    float s = 0.f;
#pragma unroll
    for (int t = 0; t < 6; t++) s += p[t] * p[t];
    sq[i] = s;
}

__global__ __launch_bounds__(256)
void dinv_kernel(const float* __restrict__ x, const float* __restrict__ sq,
                 float* __restrict__ dinv) {
    int lane = threadIdx.x & 31;
    int row  = blockIdx.x * 8 + (threadIdx.x >> 5); // 0..16383
    int b = row >> 11, i = row & 2047;
    const float* xb  = x  + (size_t)b * NPTS * 6;
    const float* sqb = sq + b * NPTS;
    float xi[6];
#pragma unroll
    for (int t = 0; t < 6; t++) xi[t] = xb[(size_t)i * 6 + t];
    float sqi = sqb[i];
    float s = 0.f;
    for (int j = lane; j < NPTS; j += 32) {
        const float* xj = xb + (size_t)j * 6;
        float dot = 0.f;
#pragma unroll
        for (int t = 0; t < 6; t++) dot += xi[t] * xj[t];
        s += __expf(-(sqi - 2.f * dot + sqb[j]));
    }
#pragma unroll
    for (int o = 16; o > 0; o >>= 1) s += __shfl_xor(s, o, 32);
    if (lane == 0) dinv[row] = rsqrtf(s);
}

__global__ void lap_kernel(const float* __restrict__ x, const float* __restrict__ sq,
                           const float* __restrict__ dinv, __bf16* __restrict__ Lap) {
    size_t idx = (size_t)blockIdx.x * 256 + threadIdx.x;  // B*N*N
    int b   = (int)(idx >> 22);
    int rem = (int)(idx & ((1u << 22) - 1));
    int i = rem >> 11, j = rem & 2047;
    const float* xb = x + (size_t)b * NPTS * 6;
    float dot = 0.f;
#pragma unroll
    for (int t = 0; t < 6; t++) dot += xb[(size_t)i * 6 + t] * xb[(size_t)j * 6 + t];
    float a = __expf(-(sq[b * NPTS + i] - 2.f * dot + sq[b * NPTS + j]));
    float v = -dinv[b * NPTS + i] * a * dinv[b * NPTS + j];
    if (i == j) v += 1.f;
    Lap[idx] = (__bf16)v;
}

// concat(x[6], onehot(cat)[16], zeros[10]) -> bf16 [16384 x 32]
__global__ void feat_kernel(const float* __restrict__ x, const int* __restrict__ cat,
                            __bf16* __restrict__ o) {
    int row = blockIdx.x * 256 + threadIdx.x;
    int b = row >> 11;
    const float* p = x + (size_t)row * 6;
    __bf16* q = o + (size_t)row * 32;
#pragma unroll
    for (int t = 0; t < 6; t++) q[t] = (__bf16)p[t];
    int c = cat[b];
#pragma unroll
    for (int t = 0; t < 16; t++) q[6 + t] = (__bf16)((t == c) ? 1.f : 0.f);
#pragma unroll
    for (int t = 22; t < 32; t++) q[t] = (__bf16)0.f;
}

// ---- weight prep: f32 [K x N] row-major -> bf16 [Npad x Kpad], zero-padded
__global__ void transpose_w(const float* __restrict__ W, int K, int N,
                            __bf16* __restrict__ Wt, int Kpad, int total) {
    int idx = blockIdx.x * 256 + threadIdx.x;
    if (idx >= total) return;
    int n = idx / Kpad;
    int k = idx - n * Kpad;
    float v = 0.f;
    if (k < K && n < N) v = W[(size_t)k * N + n];
    Wt[idx] = (__bf16)v;
}

// ----------------------- WMMA fragment helper ------------------------------
// Fragment-major LDS row (stride LST): lane's 16 bf16 = K ranges
// [half*8 .. +7] and [16+half*8 .. +7] -> two contiguous 16B ds_load_b128.

__device__ inline v16bf frag16(const __bf16* __restrict__ row, int half) {
    const v8bf lo = *(const v8bf*)(row + half * 8);
    const v8bf hi = *(const v8bf*)(row + 16 + half * 8);
    return __builtin_shufflevector(lo, hi, 0, 1, 2, 3, 4, 5, 6, 7,
                                   8, 9, 10, 11, 12, 13, 14, 15);
}

// ------------- GEMM 1: acc (+)= X @ W  (pre-transposed bf16 W) -------------
// A bf16 [M x lda], zero-padded cols; Wt bf16 [Npad x ldw] fragment-major,
// zero-padded; C f32 [M x ldc] with ldc a multiple of BN. Branch-free body;
// both tiles staged with async global->LDS copies.

__global__ __launch_bounds__(256)
void gemm_xw_kernel(const __bf16* __restrict__ A, int lda,
                    const __bf16* __restrict__ Wt, int ldw, int K,
                    float* __restrict__ C, int ldc, int accumulate) {
    __shared__ __bf16 As [BM * LST];
    __shared__ __bf16 Bst[BN * LST];
    const int tid  = threadIdx.x;
    const int lane = tid & 31;
    const int w    = tid >> 5;
    const int wm   = (w >> 1) * 32;
    const int wn   = (w & 1) * 64;
    const int fm   = lane & 15;
    const int half = lane >> 4;
    const int m0 = blockIdx.y * BM;
    const int n0 = blockIdx.x * BN;

    // staging coordinates (2 chunks of 8 bf16 per thread per tile)
    const int sr0 = tid >> 2,          sc0 = (tid & 3) * 8;
    const int sr1 = (tid + 256) >> 2,  sc1 = ((tid + 256) & 3) * 8;
    const unsigned ldsA0 = (unsigned)(size_t)&As [sr0 * LST + sc0];
    const unsigned ldsA1 = (unsigned)(size_t)&As [sr1 * LST + sc1];
    const unsigned ldsB0 = (unsigned)(size_t)&Bst[sr0 * LST + sc0];
    const unsigned ldsB1 = (unsigned)(size_t)&Bst[sr1 * LST + sc1];
    const __bf16* gA0 = &A [(size_t)(m0 + sr0) * lda + sc0];
    const __bf16* gA1 = &A [(size_t)(m0 + sr1) * lda + sc1];
    const __bf16* gB0 = &Wt[(size_t)(n0 + sr0) * ldw + sc0];
    const __bf16* gB1 = &Wt[(size_t)(n0 + sr1) * ldw + sc1];

    v8f c[2][4];
    if (accumulate) {
#pragma unroll
        for (int i = 0; i < 2; i++)
#pragma unroll
            for (int j = 0; j < 4; j++)
#pragma unroll
                for (int r = 0; r < 8; r++) {
                    int mm = m0 + wm + i * 16 + r + 8 * half;
                    int nn = n0 + wn + j * 16 + fm;
                    c[i][j][r] = C[(size_t)mm * ldc + nn];
                }
    } else {
#pragma unroll
        for (int i = 0; i < 2; i++)
#pragma unroll
            for (int j = 0; j < 4; j++)
#pragma unroll
                for (int r = 0; r < 8; r++) c[i][j][r] = 0.f;
    }

    for (int k0 = 0; k0 < K; k0 += BK) {
        ASYNC_CP16(ldsA0, gA0 + k0);
        ASYNC_CP16(ldsA1, gA1 + k0);
        ASYNC_CP16(ldsB0, gB0 + k0);
        ASYNC_CP16(ldsB1, gB1 + k0);
        if (k0 + BK < K) {   // prefetch next tiles (global_prefetch_b8)
            __builtin_prefetch(gA0 + k0 + BK, 0, 1);
            __builtin_prefetch(gB0 + k0 + BK, 0, 1);
        }
        ASYNC_DRAIN();
        __syncthreads();
        v16bf a[2], b[4];
#pragma unroll
        for (int i = 0; i < 2; i++)
            a[i] = frag16(&As[(wm + i * 16 + fm) * LST], half);
#pragma unroll
        for (int j = 0; j < 4; j++)
            b[j] = frag16(&Bst[(wn + j * 16 + fm) * LST], half);
#pragma unroll
        for (int i = 0; i < 2; i++)
#pragma unroll
            for (int j = 0; j < 4; j++)
                c[i][j] = __builtin_amdgcn_wmma_f32_16x16x32_bf16(
                    false, a[i], false, b[j], (short)0, c[i][j], false, false);
        __syncthreads();
    }

#pragma unroll
    for (int i = 0; i < 2; i++)
#pragma unroll
        for (int j = 0; j < 4; j++)
#pragma unroll
            for (int r = 0; r < 8; r++) {
                int mm = m0 + wm + i * 16 + r + 8 * half;
                int nn = n0 + wn + j * 16 + fm;
                C[(size_t)mm * ldc + nn] = c[i][j][r];
            }
}

// ------ GEMM 2: Chebyshev step  Out = bf16( alpha*(L @ X) - X0 )  ----------
// Batched via blockIdx.z; M = K = 2048. In-place Out==X0 is safe (each
// element read exactly once, by the thread that overwrites it). B staging
// reads unconditional 16B chunks; columns >= Nn carry garbage that only feeds
// output columns >= Nn, which are never stored (and stay inside d_ws).

__global__ __launch_bounds__(256)
void gemm_lx_kernel(const __bf16* __restrict__ Lap, const __bf16* __restrict__ X,
                    const __bf16* __restrict__ X0, __bf16* __restrict__ Out,
                    int ldx, int Nn, float alpha) {
    __shared__ __bf16 As [BM * LST];
    __shared__ __bf16 Bst[BN * LST];
    const int tid  = threadIdx.x;
    const int lane = tid & 31;
    const int w    = tid >> 5;
    const int wm   = (w >> 1) * 32;
    const int wn   = (w & 1) * 64;
    const int fm   = lane & 15;
    const int half = lane >> 4;
    const int m0 = blockIdx.y * BM;
    const int n0 = blockIdx.x * BN;
    const __bf16* Ab  = Lap + (size_t)blockIdx.z * NPTS * NPTS;
    const __bf16* Bb  = X   + (size_t)blockIdx.z * NPTS * ldx;
    const __bf16* X0b = X0 ? (X0 + (size_t)blockIdx.z * NPTS * ldx) : nullptr;
    __bf16* Ob = Out + (size_t)blockIdx.z * NPTS * ldx;

    const int sr0 = tid >> 2,          sc0 = (tid & 3) * 8;
    const int sr1 = (tid + 256) >> 2,  sc1 = ((tid + 256) & 3) * 8;
    const unsigned ldsA0 = (unsigned)(size_t)&As[sr0 * LST + sc0];
    const unsigned ldsA1 = (unsigned)(size_t)&As[sr1 * LST + sc1];
    const __bf16* gA0 = &Ab[(size_t)(m0 + sr0) * NPTS + sc0];
    const __bf16* gA1 = &Ab[(size_t)(m0 + sr1) * NPTS + sc1];

    v8f c[2][4];
#pragma unroll
    for (int i = 0; i < 2; i++)
#pragma unroll
        for (int j = 0; j < 4; j++)
#pragma unroll
            for (int r = 0; r < 8; r++) c[i][j][r] = 0.f;

    for (int k0 = 0; k0 < NPTS; k0 += BK) {
        // L tile: async global -> LDS (16B per lane)
        ASYNC_CP16(ldsA0, gA0 + k0);
        ASYNC_CP16(ldsA1, gA1 + k0);
        // X tile: 32x128 bf16, vector load then transpose-scatter to N-major
#pragma unroll
        for (int g = 0; g < 2; g++) {
            int idx = tid + g * 256;
            int kr = idx >> 4, co = (idx & 15) * 8;
            v8bf v = *(const v8bf*)&Bb[(size_t)(k0 + kr) * ldx + n0 + co];
#pragma unroll
            for (int t = 0; t < 8; t++)
                Bst[(co + t) * LST + kr] = v[t];
        }
        if (k0 + BK < NPTS) {   // prefetch next L tile
            __builtin_prefetch(gA0 + k0 + BK, 0, 1);
        }
        ASYNC_DRAIN();
        __syncthreads();
        v16bf a[2], b[4];
#pragma unroll
        for (int i = 0; i < 2; i++)
            a[i] = frag16(&As[(wm + i * 16 + fm) * LST], half);
#pragma unroll
        for (int j = 0; j < 4; j++)
            b[j] = frag16(&Bst[(wn + j * 16 + fm) * LST], half);
#pragma unroll
        for (int i = 0; i < 2; i++)
#pragma unroll
            for (int j = 0; j < 4; j++)
                c[i][j] = __builtin_amdgcn_wmma_f32_16x16x32_bf16(
                    false, a[i], false, b[j], (short)0, c[i][j], false, false);
        __syncthreads();
    }

#pragma unroll
    for (int i = 0; i < 2; i++)
#pragma unroll
        for (int j = 0; j < 4; j++)
#pragma unroll
            for (int r = 0; r < 8; r++) {
                int mm = m0 + wm + i * 16 + r + 8 * half;
                int nn = n0 + wn + j * 16 + fm;
                if (nn < Nn) {
                    float v = alpha * c[i][j][r];
                    if (X0b) v -= (float)X0b[(size_t)mm * ldx + nn];
                    Ob[(size_t)mm * ldx + nn] = (__bf16)v;
                }
            }
}

// --------------------------- epilogues -------------------------------------

__global__ void epi_bf16(const float* __restrict__ acc, int ldc, int F,
                         const float* __restrict__ bias, const float* __restrict__ br,
                         __bf16* __restrict__ o, int total) {
    int i = blockIdx.x * 256 + threadIdx.x;
    if (i >= total) return;
    int row = i / F;
    int f   = i - row * F;
    int n   = row & 2047;
    float v = acc[(size_t)row * ldc + f] + bias[f] + br[n * F + f];
    o[i] = (__bf16)(v > 0.f ? v : 0.f);
}

__global__ void epi_f32(const float* __restrict__ acc, int ldc, int F,
                        const float* __restrict__ bias, const float* __restrict__ br,
                        float* __restrict__ o, int total) {
    int i = blockIdx.x * 256 + threadIdx.x;
    if (i >= total) return;
    int row = i / F;
    int f   = i - row * F;
    int n   = row & 2047;
    float v = acc[(size_t)row * ldc + f] + bias[f] + br[n * F + f];
    o[i] = (v > 0.f ? v : 0.f);
}

// ----------------------------- orchestration -------------------------------

extern "C" void kernel_launch(void* const* d_in, const int* in_sizes, int n_in,
                              void* d_out, int out_size, void* d_ws, size_t ws_size,
                              hipStream_t stream) {
    const float* x   = (const float*)d_in[0];
    const int*   cat = (const int*)d_in[1];
    const float* cw0 = (const float*)d_in[2];
    const float* cb0 = (const float*)d_in[3];
    const float* cw1 = (const float*)d_in[4];
    const float* cb1 = (const float*)d_in[5];
    const float* cw2 = (const float*)d_in[6];
    const float* cb2 = (const float*)d_in[7];
    const float* fw0 = (const float*)d_in[8];
    const float* fb0 = (const float*)d_in[9];
    const float* fw1 = (const float*)d_in[10];
    const float* fb1 = (const float*)d_in[11];
    const float* fw2 = (const float*)d_in[12];
    const float* fb2 = (const float*)d_in[13];
    const float* br0 = (const float*)d_in[14];
    const float* br1 = (const float*)d_in[15];
    const float* br2 = (const float*)d_in[16];
    const float* br3 = (const float*)d_in[17];
    const float* br4 = (const float*)d_in[18];
    const float* br5 = (const float*)d_in[19];
    float* out = (float*)d_out;

    // workspace carve-up (~241 MB)
    char* ws = (char*)d_ws;
    __bf16* Lap  = (__bf16*)(ws);                              // 64MB
    float*  acc  = (float*)(ws + (size_t)67108864);            // 64MB
    __bf16* act0 = (__bf16*)(ws + (size_t)134217728);          // 32MB
    __bf16* act1 = (__bf16*)(ws + (size_t)167772160);          // 32MB (x1 skip)
    __bf16* t1   = (__bf16*)(ws + (size_t)201326592);          // 16MB
    __bf16* t2   = (__bf16*)(ws + (size_t)218103808);          // 16MB
    float*  sq   = (float*)(ws + (size_t)234881024);           // 64KB
    float*  dinv = (float*)(ws + (size_t)234946560);           // 64KB
    __bf16* wb   = (__bf16*)(ws + (size_t)235012096);          // ~5.2MB bf16 weights
    __bf16* cw0t = wb;                          // 6 x [128][32]
    __bf16* cw1t = cw0t + 6 * 4096;             // 5 x [512][128]
    __bf16* cw2t = cw1t + 5 * 65536;            // 3 x [1024][512]
    __bf16* fw0t = cw2t + 3 * 524288;           // [512][1024]
    __bf16* fw1t = fw0t + 524288;               // [128][1024]
    __bf16* fw2t = fw1t + 131072;               // [128][128] (N padded 50->128)

    // --- weight prep (bf16, transposed, padded) ---
    auto TW = [&](const float* W, int K, int N, __bf16* Wt, int Kpad, int Npad) {
        int total = Npad * Kpad;
        transpose_w<<<(total + 255) / 256, 256, 0, stream>>>(W, K, N, Wt, Kpad, total);
    };
    for (int k = 0; k < 6; k++) TW(cw0 + k * 22 * 128,   22,  128,  cw0t + k * 4096,   32,   128);
    for (int k = 0; k < 5; k++) TW(cw1 + k * 128 * 512,  128, 512,  cw1t + k * 65536,  128,  512);
    for (int k = 0; k < 3; k++) TW(cw2 + k * 512 * 1024, 512, 1024, cw2t + k * 524288, 512,  1024);
    TW(fw0, 1024, 512, fw0t, 1024, 512);
    TW(fw1, 1024, 128, fw1t, 1024, 128);
    TW(fw2, 128,  50,  fw2t, 128,  128);

    // --- Laplacian + input features ---
    sq_kernel  <<<ROWS / 256, 256, 0, stream>>>(x, sq);
    dinv_kernel<<<ROWS / 8,   256, 0, stream>>>(x, sq, dinv);
    lap_kernel <<<(unsigned)(((size_t)NB * NPTS * NPTS) / 256), 256, 0, stream>>>(x, sq, dinv, Lap);
    feat_kernel<<<ROWS / 256, 256, 0, stream>>>(x, cat, act0);

    auto XW = [&](const __bf16* A, int lda, int K, const __bf16* Wt, int ldw,
                  int Npad, int accum) {
        dim3 g(Npad / BN, ROWS / BM, 1);
        gemm_xw_kernel<<<g, 256, 0, stream>>>(A, lda, Wt, ldw, K, acc, Npad, accum);
    };
    auto LX = [&](const __bf16* X, const __bf16* X0, __bf16* O, int width, float alpha) {
        dim3 g((width + BN - 1) / BN, NPTS / BM, NB);
        gemm_lx_kernel<<<g, 256, 0, stream>>>(Lap, X, X0, O, width, width, alpha);
    };
    auto EPI = [&](int F, int ldc, const float* bias, const float* brp, __bf16* o) {
        int total = ROWS * F;
        epi_bf16<<<(total + 255) / 256, 256, 0, stream>>>(acc, ldc, F, bias, brp, o, total);
    };

    // --- Cheb layer 0: K=6, 22 -> 128 (features stored padded to 32) ---
    XW(act0, 32, 32, cw0t + 0 * 4096, 32, 128, 0);
    LX(act0, nullptr, t1, 32, 1.f);
    XW(t1, 32, 32, cw0t + 1 * 4096, 32, 128, 1);
    LX(t1, act0, t2, 32, 2.f);  XW(t2, 32, 32, cw0t + 2 * 4096, 32, 128, 1);
    LX(t2, t1,   t1, 32, 2.f);  XW(t1, 32, 32, cw0t + 3 * 4096, 32, 128, 1);
    LX(t1, t2,   t2, 32, 2.f);  XW(t2, 32, 32, cw0t + 4 * 4096, 32, 128, 1);
    LX(t2, t1,   t1, 32, 2.f);  XW(t1, 32, 32, cw0t + 5 * 4096, 32, 128, 1);
    EPI(128, 128, cb0, br0, act0);

    // --- Cheb layer 1: K=5, 128 -> 512 ---
    XW(act0, 128, 128, cw1t + 0 * 65536, 128, 512, 0);
    LX(act0, nullptr, t1, 128, 1.f);
    XW(t1, 128, 128, cw1t + 1 * 65536, 128, 512, 1);
    LX(t1, act0, t2, 128, 2.f); XW(t2, 128, 128, cw1t + 2 * 65536, 128, 512, 1);
    LX(t2, t1,   t1, 128, 2.f); XW(t1, 128, 128, cw1t + 3 * 65536, 128, 512, 1);
    LX(t1, t2,   t2, 128, 2.f); XW(t2, 128, 128, cw1t + 4 * 65536, 128, 512, 1);
    EPI(512, 512, cb1, br1, act1);      // x1 skip connection lives in act1

    // --- Cheb layer 2: K=3, 512 -> 1024 ---
    XW(act1, 512, 512, cw2t + 0 * 524288, 512, 1024, 0);
    LX(act1, nullptr, t1, 512, 1.f);
    XW(t1, 512, 512, cw2t + 1 * 524288, 512, 1024, 1);
    LX(t1, act1, t2, 512, 2.f);
    XW(t2, 512, 512, cw2t + 2 * 524288, 512, 1024, 1);
    EPI(1024, 1024, cb2, br2, act0);

    // --- fc0: 1024 -> 512 ---
    XW(act0, 1024, 1024, fw0t, 1024, 512, 0);
    EPI(512, 512, fb0, br3, act0);

    // --- fc1: concat(fc0_out, x1) 1024 -> 128 (two GEMMs over split K) ---
    XW(act0, 512, 512, fw1t,       1024, 128, 0);
    XW(act1, 512, 512, fw1t + 512, 1024, 128, 1);
    EPI(128, 128, fb1, br4, act0);

    // --- fc2: 128 -> 50 (N padded to 128 in acc), f32 output ---
    XW(act0, 128, 128, fw2t, 128, 128, 0);
    {
        int total = ROWS * 50;
        epi_f32<<<(total + 255) / 256, 256, 0, stream>>>(acc, 128, 50, fb2, br5, out, total);
    }
}
